// SelfAttention_9895604650571
// MI455X (gfx1250) — matrix operand order repaired and checked by reading, not compiled, not run on hardware
//
#include <hip/hip_runtime.h>
#include <hip/hip_bf16.h>

// MI455X / gfx1250 self-attention, all matmuls on v_wmma_f32_16x16x32_bf16.
// B=2, S=2048, D=1024, H=16, HD=64.
// Layouts chosen so every WMMA fragment is two contiguous 128-bit LDS loads.

typedef __attribute__((ext_vector_type(16))) __bf16          v16bf;
typedef __attribute__((ext_vector_type(16))) unsigned short  v16u;
typedef __attribute__((ext_vector_type(8)))  unsigned short  v8u;
typedef __attribute__((ext_vector_type(8)))  float           v8f;

union frag16 { v16u v; v8u h[2]; };

__device__ __forceinline__ unsigned short f32_to_bf16_rne(float f) {
    unsigned int u = __float_as_uint(f);
    unsigned int r = u + 0x7FFFu + ((u >> 16) & 1u);
    return (unsigned short)(r >> 16);
}

__device__ __forceinline__ v8f wmma_bf16(v16u a, v16u b, v8f c) {
    return __builtin_amdgcn_wmma_f32_16x16x32_bf16(
        false, __builtin_bit_cast(v16bf, a),
        false, __builtin_bit_cast(v16bf, b),
        (short)0, c, false, false);
}

// ---------------------------------------------------------------- convert
__global__ void cvt_f32_bf16(const float* __restrict__ in,
                             unsigned short* __restrict__ out, int n) {
    int i = blockIdx.x * blockDim.x + threadIdx.x;
    if (i < n) out[i] = f32_to_bf16_rne(in[i]);
}

// out[n*K + k] = bf16(in[k*N + n])  (weight transpose, 32x32 LDS tiles)
__global__ __launch_bounds__(256) void cvt_transpose_bf16(
    const float* __restrict__ in, unsigned short* __restrict__ out,
    int K, int N) {
    __shared__ float tile[32][33];
    const int k0 = blockIdx.y * 32, n0 = blockIdx.x * 32;
    const int tx = threadIdx.x, ty = threadIdx.y;      // 32 x 8
    #pragma unroll
    for (int j = 0; j < 32; j += 8)
        tile[ty + j][tx] = in[(size_t)(k0 + ty + j) * N + n0 + tx];
    __syncthreads();
    #pragma unroll
    for (int j = 0; j < 32; j += 8)
        out[(size_t)(n0 + ty + j) * K + k0 + tx] = f32_to_bf16_rne(tile[tx][ty + j]);
}

// ---------------------------------------------------------------- GEMM
// C[M,N] = A[M,K](bf16 row-major) @ Wt[N,K](bf16, pre-transposed) + bias.
// Block tile 64x64, 128 threads = 4 wave32s, K stepped by 32.
// MODE: 0 = f32 row-major out, 1 = bf16 row-major out,
//       2 = bf16 column-major out (out[n*M + m], vectorized b128 stores).
template <int MODE>
__global__ __launch_bounds__(128) void gemm_bf16_k(
    const unsigned short* __restrict__ A,
    const unsigned short* __restrict__ Wt,
    const float* __restrict__ bias,
    void* __restrict__ out,
    int M, int N, int K)
{
    __shared__ __align__(16) unsigned short As[64][40];  // 64x32 (+16B-mult pad)
    __shared__ __align__(16) unsigned short Bt[64][40];  // 64(n) x 32(k)

    const int tid  = threadIdx.x;
    const int wave = tid >> 5;
    const int lane = tid & 31;
    const int l16  = lane & 15;
    const int half = lane >> 4;
    const int tileM = blockIdx.y * 64;
    const int tileN = blockIdx.x * 64;

    const v8f z = {0.f,0.f,0.f,0.f,0.f,0.f,0.f,0.f};
    v8f acc[4] = {z, z, z, z};

    for (int kc = 0; kc < K; kc += 32) {
        // 64 rows x 4 v8u chunks, both tiles: 2 chunk-copies per thread each.
        #pragma unroll
        for (int i = tid; i < 256; i += 128) {
            const int r = i >> 2, c8 = (i & 3) * 8;
            *(v8u*)&As[r][c8] = *(const v8u*)&A [(size_t)(tileM + r) * K + kc + c8];
        }
        #pragma unroll
        for (int i = tid; i < 256; i += 128) {
            const int r = i >> 2, c8 = (i & 3) * 8;
            *(v8u*)&Bt[r][c8] = *(const v8u*)&Wt[(size_t)(tileN + r) * K + kc + c8];
        }
        __syncthreads();

        // A fragment: elems 0..7 = K=half*8.., elems 8..15 = K=16+half*8..
        const int m = wave * 16 + l16;
        frag16 af;
        af.h[0] = *(const v8u*)&As[m][half * 8];
        af.h[1] = *(const v8u*)&As[m][16 + half * 8];
        #pragma unroll
        for (int t = 0; t < 4; ++t) {
            // B fragment: lane = col n; elems 0..7 = K=half*8.., elems 8..15 = K=16+half*8..
            const int n = t * 16 + l16;
            frag16 bf;
            bf.h[0] = *(const v8u*)&Bt[n][half * 8];
            bf.h[1] = *(const v8u*)&Bt[n][16 + half * 8];
            acc[t] = wmma_bf16(af.v, bf.v, acc[t]);
        }
        __syncthreads();
    }

    // C/D layout: VGPR r -> M = r + 8*half (8 consecutive rows), N = lane%16.
    #pragma unroll
    for (int t = 0; t < 4; ++t) {
        const int n  = tileN + t * 16 + l16;
        const float bv = bias[n];
        const int mg0 = tileM + wave * 16 + half * 8;
        if (MODE == 2) {
            v8u pk;
            #pragma unroll
            for (int r = 0; r < 8; ++r) pk[r] = f32_to_bf16_rne(acc[t][r] + bv);
            *(v8u*)&((unsigned short*)out)[(size_t)n * M + mg0] = pk;  // b128 store
        } else {
            #pragma unroll
            for (int r = 0; r < 8; ++r) {
                const float v = acc[t][r] + bv;
                if (MODE == 1)
                    ((unsigned short*)out)[(size_t)(mg0 + r) * N + n] = f32_to_bf16_rne(v);
                else
                    ((float*)out)[(size_t)(mg0 + r) * N + n] = v;
            }
        }
    }
}

// ---------------------------------------------------------------- attention
// One block = 64 query rows of one (b,h). 128 threads = 4 waves.
// q,k: bf16 row-major (B*S, D).  vt: bf16 column-major (D, B*S):
//   vt[(h*64+hd)*M + b*S + s]  -> V^T tiles load/read fully vectorized.
__global__ __launch_bounds__(128) void attn_k(
    const unsigned short* __restrict__ qb,
    const unsigned short* __restrict__ kb,
    const unsigned short* __restrict__ vt,
    unsigned short* __restrict__ ctx,
    int S, int D, int H, int M)
{
    __shared__ __align__(16) unsigned short Qs [64][72];
    __shared__ __align__(16) unsigned short Ks [64][72];
    __shared__ __align__(16) unsigned short VsT[64][72];  // [hd][key]
    __shared__ __align__(16) unsigned short Ps [64][72];
    __shared__ float Sc[64][65];
    __shared__ float row_m[64], row_l[64], row_a[64];

    const int b  = blockIdx.y / H;
    const int h  = blockIdx.y % H;
    const int q0 = blockIdx.x * 64;
    const int tid  = threadIdx.x;
    const int wave = tid >> 5;
    const int lane = tid & 31;
    const int l16  = lane & 15;
    const int half = lane >> 4;
    const int m    = wave * 16 + l16;

    // Q tile: 64x64, vectorized (512 v8u chunks / 128 threads).
    #pragma unroll
    for (int i = tid; i < 512; i += 128) {
        const int r = i >> 3, c8 = (i & 7) * 8;
        *(v8u*)&Qs[r][c8] = *(const v8u*)&qb[(size_t)(b * S + q0 + r) * D + h * 64 + c8];
    }
    if (tid < 64) { row_m[tid] = -1e30f; row_l[tid] = 0.f; }

    const v8f z = {0.f,0.f,0.f,0.f,0.f,0.f,0.f,0.f};
    v8f o[4] = {z, z, z, z};
    const float scale = 0.125f;   // 1/sqrt(64)
    __syncthreads();

    // Loop-invariant Q fragments.
    frag16 qf[2];
    #pragma unroll
    for (int ko = 0; ko < 2; ++ko) {
        qf[ko].h[0] = *(const v8u*)&Qs[m][ko * 32 + half * 8];
        qf[ko].h[1] = *(const v8u*)&Qs[m][ko * 32 + 16 + half * 8];
    }

    for (int jb = 0; jb <= (int)blockIdx.x; ++jb) {
        const int k0 = jb * 64;
        #pragma unroll
        for (int i = tid; i < 512; i += 128) {
            const int r = i >> 3, c8 = (i & 7) * 8;
            *(v8u*)&Ks[r][c8]  = *(const v8u*)&kb[(size_t)(b * S + k0 + r) * D + h * 64 + c8];
            *(v8u*)&VsT[r][c8] = *(const v8u*)&vt[(size_t)(h * 64 + r) * M + b * S + k0 + c8];
        }
        __syncthreads();

        // S = Q @ K^T : B[k=hd][n=key] = Ks[key][hd]  (row-contiguous).
        #pragma unroll
        for (int t = 0; t < 4; ++t) {
            const int n = t * 16 + l16;
            v8f s = z;
            #pragma unroll
            for (int ko = 0; ko < 2; ++ko) {
                frag16 kf;
                kf.h[0] = *(const v8u*)&Ks[n][ko * 32 + half * 8];
                kf.h[1] = *(const v8u*)&Ks[n][ko * 32 + 16 + half * 8];
                s = wmma_bf16(qf[ko].v, kf.v, s);
            }
            const int kpos = k0 + n;
            #pragma unroll
            for (int r = 0; r < 8; ++r) {
                const int qrow = wave * 16 + half * 8 + r;
                float v = s[r] * scale;
                if (kpos > q0 + qrow) v = -1e30f;   // causal mask
                Sc[qrow][n] = v;
            }
        }
        __syncthreads();

        // Online softmax per row (64 rows, one thread each).
        if (tid < 64) {
            const int r = tid;
            float mo = row_m[r], mx = mo;
            #pragma unroll 8
            for (int j = 0; j < 64; ++j) mx = fmaxf(mx, Sc[r][j]);
            const float alpha = __expf(mo - mx);
            float sum = 0.f;
            #pragma unroll 8
            for (int j = 0; j < 64; ++j) {
                const float p = __expf(Sc[r][j] - mx);
                sum += p;
                Ps[r][j] = f32_to_bf16_rne(p);
            }
            row_m[r] = mx;
            row_l[r] = row_l[r] * alpha + sum;
            row_a[r] = alpha;
        }
        __syncthreads();

        // Rescale running output, then O += P @ V.
        #pragma unroll
        for (int t = 0; t < 4; ++t)
            #pragma unroll
            for (int r = 0; r < 8; ++r)
                o[t][r] *= row_a[wave * 16 + half * 8 + r];

        frag16 pf[2];
        #pragma unroll
        for (int ko = 0; ko < 2; ++ko) {
            pf[ko].h[0] = *(const v8u*)&Ps[m][ko * 32 + half * 8];
            pf[ko].h[1] = *(const v8u*)&Ps[m][ko * 32 + 16 + half * 8];
        }
        #pragma unroll
        for (int t = 0; t < 4; ++t) {
            const int n = t * 16 + l16;           // hd column
            #pragma unroll
            for (int ko = 0; ko < 2; ++ko) {
                frag16 vf;                         // B[k=key][n=hd] = VsT[hd][key]
                vf.h[0] = *(const v8u*)&VsT[n][ko * 32 + half * 8];
                vf.h[1] = *(const v8u*)&VsT[n][ko * 32 + 16 + half * 8];
                o[t] = wmma_bf16(pf[ko].v, vf.v, o[t]);
            }
        }
        __syncthreads();
    }

    // Normalize and store bf16 ctx row-major (consumed as GEMM A matrix).
    #pragma unroll
    for (int t = 0; t < 4; ++t) {
        const int n = t * 16 + l16;
        #pragma unroll
        for (int r = 0; r < 8; ++r) {
            const int ml = wave * 16 + half * 8 + r;
            const float v = o[t][r] / row_l[ml];
            ctx[(size_t)(b * S + q0 + ml) * D + h * 64 + n] = f32_to_bf16_rne(v);
        }
    }
}

// ---------------------------------------------------------------- launcher
extern "C" void kernel_launch(void* const* d_in, const int* in_sizes, int n_in,
                              void* d_out, int out_size, void* d_ws, size_t ws_size,
                              hipStream_t stream) {
    (void)in_sizes; (void)n_in; (void)out_size; (void)ws_size;

    const int B = 2, S = 2048, D = 1024, H = 16;
    const int M = B * S;                 // 4096 token rows

    const float* x  = (const float*)d_in[0];
    const float* Wq = (const float*)d_in[1];
    const float* bq = (const float*)d_in[2];
    const float* Wk = (const float*)d_in[3];
    const float* bk = (const float*)d_in[4];
    const float* Wv = (const float*)d_in[5];
    const float* bv = (const float*)d_in[6];
    const float* Wo = (const float*)d_in[7];
    const float* bo = (const float*)d_in[8];

    // bf16 staging in workspace (48 MB total).
    unsigned short* xb   = (unsigned short*)d_ws;
    unsigned short* Wqt  = xb  + (size_t)M * D;     // weights stored transposed (N x K)
    unsigned short* Wkt  = Wqt + (size_t)D * D;
    unsigned short* Wvt  = Wkt + (size_t)D * D;
    unsigned short* Wot  = Wvt + (size_t)D * D;
    unsigned short* qbuf = Wot + (size_t)D * D;     // row-major
    unsigned short* kbuf = qbuf + (size_t)M * D;    // row-major
    unsigned short* vtbf = kbuf + (size_t)M * D;    // column-major (D x M)
    unsigned short* cbuf = vtbf + (size_t)M * D;    // row-major

    const int nx = M * D;
    cvt_f32_bf16<<<(nx + 255) / 256, 256, 0, stream>>>(x, xb, nx);
    dim3 tblk(32, 8);
    dim3 tgrid(D / 32, D / 32);
    cvt_transpose_bf16<<<tgrid, tblk, 0, stream>>>(Wq, Wqt, D, D);
    cvt_transpose_bf16<<<tgrid, tblk, 0, stream>>>(Wk, Wkt, D, D);
    cvt_transpose_bf16<<<tgrid, tblk, 0, stream>>>(Wv, Wvt, D, D);
    cvt_transpose_bf16<<<tgrid, tblk, 0, stream>>>(Wo, Wot, D, D);

    dim3 gblk(128);
    dim3 ggrid(D / 64, M / 64);          // (16, 64)
    gemm_bf16_k<1><<<ggrid, gblk, 0, stream>>>(xb, Wqt, bq, qbuf, M, D, D);
    gemm_bf16_k<1><<<ggrid, gblk, 0, stream>>>(xb, Wkt, bk, kbuf, M, D, D);
    gemm_bf16_k<2><<<ggrid, gblk, 0, stream>>>(xb, Wvt, bv, vtbf, M, D, D);

    dim3 agrid(S / 64, B * H);           // (32, 32)
    attn_k<<<agrid, gblk, 0, stream>>>(qbuf, kbuf, vtbf, cbuf, S, D, H, M);

    gemm_bf16_k<0><<<ggrid, gblk, 0, stream>>>(cbuf, Wot, bo, d_out, M, D, D);
}
